// PointNet_18442589570241
// MI455X (gfx1250) — matrix operand order repaired and checked by reading, not compiled, not run on hardware
//
#include <hip/hip_runtime.h>
#include <hip/hip_bf16.h>

typedef _Float16 h16;
typedef __attribute__((ext_vector_type(16))) _Float16 v16h;
typedef __attribute__((ext_vector_type(8)))  _Float16 v8h;
typedef __attribute__((ext_vector_type(8)))  float    v8f;

// ---------------------------------------------------------------------------
// f32 -> f16 weight convert with K zero-padding. out [Nr, Kp], in [Nr, K].
// ---------------------------------------------------------------------------
__global__ void k_wpad(const float* __restrict__ in, h16* __restrict__ out,
                       int Nr, int K, int Kp) {
  int t = blockIdx.x * blockDim.x + threadIdx.x;
  int total = Nr * Kp;
  if (t >= total) return;
  int r = t / Kp, k = t % Kp;
  out[t] = (k < K) ? (h16)in[(size_t)r * K + k] : (h16)0.0f;
}

// ---------------------------------------------------------------------------
// Farthest point sampling: one block per batch, dist kept in registers.
// sel[0]=0; dist updated with previous selection before each argmax
// (matches the lax.scan semantics in the reference).
// ---------------------------------------------------------------------------
template <int PPT>
__global__ void k_fps(const float* __restrict__ xyz, int n, int npoint,
                      float* __restrict__ new_xyz) {
  const int b = blockIdx.x;
  const int t = threadIdx.x;
  const int T = blockDim.x;
  const float* p = xyz + (size_t)b * n * 3;

  float px[PPT], py[PPT], pz[PPT], dist[PPT];
#pragma unroll
  for (int j = 0; j < PPT; ++j) {
    int i = t + j * T;
    if (i < n) {
      px[j] = p[i * 3 + 0]; py[j] = p[i * 3 + 1]; pz[j] = p[i * 3 + 2];
      dist[j] = 1e10f;
    } else {
      px[j] = py[j] = pz[j] = 0.0f;
      dist[j] = -1e30f;
    }
  }

  __shared__ float sval[1024];
  __shared__ int   sidx[1024];
  __shared__ float cent[3];

  int far = 0;
  for (int it = 0; it < npoint; ++it) {
    if (t == 0) {
      const float* c = p + (size_t)far * 3;
      float* o = new_xyz + ((size_t)b * npoint + it) * 3;
      cent[0] = c[0]; cent[1] = c[1]; cent[2] = c[2];
      o[0] = c[0]; o[1] = c[1]; o[2] = c[2];
    }
    __syncthreads();
    float cx = cent[0], cy = cent[1], cz = cent[2];
    float best = -1e30f; int bi = 0;
#pragma unroll
    for (int j = 0; j < PPT; ++j) {
      int i = t + j * T;
      if (i < n) {
        float dx = px[j] - cx, dy = py[j] - cy, dz = pz[j] - cz;
        float d = dx * dx + dy * dy + dz * dz;
        if (d < dist[j]) dist[j] = d;
        if (dist[j] > best) { best = dist[j]; bi = i; }
      }
    }
    sval[t] = best; sidx[t] = bi;
    __syncthreads();
    for (int s = T >> 1; s > 0; s >>= 1) {
      if (t < s) {
        float ov = sval[t + s]; int oi = sidx[t + s];
        if (ov > sval[t] || (ov == sval[t] && oi < sidx[t])) {
          sval[t] = ov; sidx[t] = oi;
        }
      }
      __syncthreads();
    }
    far = sidx[0];
    __syncthreads();
  }
}

// ---------------------------------------------------------------------------
// Ball query: one wave32 per center; ordered compaction via ballot matches
// "mark OOR as n, sort, take first K, pad with first hit".
// ---------------------------------------------------------------------------
__global__ void k_ballquery(const float* __restrict__ xyz,
                            const float* __restrict__ new_xyz,
                            int* __restrict__ gidx,
                            int n, int S, int K, float r2) {
  int lane = threadIdx.x & 31;
  int c = blockIdx.x;  // b*S + s
  const float* ctr = new_xyz + (size_t)c * 3;
  float cx = ctr[0], cy = ctr[1], cz = ctr[2];
  int b = c / S;
  const float* p = xyz + (size_t)b * n * 3;
  int* out = gidx + (size_t)c * K;

  int count = 0, first = -1;
  for (int base = 0; base < n && count < K; base += 32) {
    int i = base + lane;
    bool inr = false;
    if (i < n) {
      float dx = p[i * 3 + 0] - cx, dy = p[i * 3 + 1] - cy, dz = p[i * 3 + 2] - cz;
      inr = (dx * dx + dy * dy + dz * dz) <= r2;
    }
    unsigned m = (unsigned)__ballot(inr);
    if (first < 0 && m) first = base + __builtin_ctz(m);
    if (inr) {
      int pos = count + __popc(m & ((1u << lane) - 1u));
      if (pos < K) out[pos] = i;
    }
    count += __popc(m);
  }
  if (first < 0) first = 0;
  if (count > K) count = K;
  for (int j = count + lane; j < K; j += 32) out[j] = first;
}

// ---------------------------------------------------------------------------
// Grouped feature build for SA stages: row r -> [gxyz(3) | pts(Cp) | 0-pad].
// ---------------------------------------------------------------------------
__global__ void k_group_feat(const float* __restrict__ xyz, int n,
                             const float* __restrict__ pts, int np, int Cp,
                             const float* __restrict__ new_xyz,
                             const int* __restrict__ gidx,
                             int S, int K, h16* __restrict__ out, int ldk, int Rtot) {
  int r = blockIdx.x * blockDim.x + threadIdx.x;
  if (r >= Rtot) return;
  int sc = r / K;              // b*S + s
  int b = sc / S;
  int idx = gidx[r];
  if (idx >= n) idx = n - 1;
  if (idx < 0) idx = 0;
  const float* ctr = new_xyz + (size_t)sc * 3;
  const float* pp = xyz + ((size_t)b * n + idx) * 3;
  h16* o = out + (size_t)r * ldk;
  o[0] = (h16)(pp[0] - ctr[0]);
  o[1] = (h16)(pp[1] - ctr[1]);
  o[2] = (h16)(pp[2] - ctr[2]);
  const float* f = pts + ((size_t)b * np + idx) * Cp;
  for (int c = 0; c < Cp; ++c) o[3 + c] = (h16)f[c];
  for (int c = 3 + Cp; c < ldk; ++c) o[c] = (h16)0.0f;
}

// ---------------------------------------------------------------------------
// WMMA GEMM: C[R,N] = act(A[R,KP] * W[N,KP]^T + bias).
// One wave computes a 16x64 tile (4 N-subtiles share one A fragment per
// K-step -> 4 v_wmma per step, A traffic /4). Fragments are gathered with
// 16-byte global_load_b128s following the CDNA5 ISA 7.12.2 layouts:
//   A 16x32 f16 : lane m=lane&15, two contiguous 8-half runs
//                 [kb+half*8 .. +7] and [kb+16+half*8 .. +7]
//   B 32x16 f16 : lane n=lane&15, two contiguous 8-half runs
//                 [kb+half*8 .. +7] and [kb+16+half*8 .. +7]  (col n of B == row n of W)
//   C/D f32     : m=(lane>>4)*8+v, n=lane&15
// ---------------------------------------------------------------------------
union V16U { v16h v; v8h h[2]; };

template <int KP>
__global__ void k_gemm(const h16* __restrict__ A, const h16* __restrict__ W,
                       const float* __restrict__ bias,
                       float* __restrict__ C32, h16* __restrict__ C16,
                       int ldc, int R, int relu) {
  const int lane = threadIdx.x & 31;
  const int tM = blockIdx.x, tN = blockIdx.y;  // tN covers 64 output cols
  const int half_sel = lane >> 4;
  const int l15 = lane & 15;
  const int m_glob = tM * 16 + l15;
  const bool mvalid = (m_glob < R);
  const h16* Arow = A + (size_t)m_glob * KP + half_sel * 8;

  const h16* Wrow[4];
#pragma unroll
  for (int nt = 0; nt < 4; ++nt)
    Wrow[nt] = W + (size_t)(tN * 64 + nt * 16 + l15) * KP + half_sel * 8;

  v8f acc[4];
#pragma unroll
  for (int nt = 0; nt < 4; ++nt)
    acc[nt] = (v8f){0.f, 0.f, 0.f, 0.f, 0.f, 0.f, 0.f, 0.f};

  const v8h zero8 = {};

  for (int kb = 0; kb < KP; kb += 32) {
    V16U a;
    a.h[0] = mvalid ? *(const v8h*)(Arow + kb)      : zero8;
    a.h[1] = mvalid ? *(const v8h*)(Arow + kb + 16) : zero8;
    if (mvalid) __builtin_prefetch(Arow + kb + 32, 0, 1);  // global_prefetch_b8
#pragma unroll
    for (int nt = 0; nt < 4; ++nt) {
      V16U bf;
      bf.h[0] = *(const v8h*)(Wrow[nt] + kb);
      bf.h[1] = *(const v8h*)(Wrow[nt] + kb + 16);
      acc[nt] = __builtin_amdgcn_wmma_f32_16x16x32_f16(false, a.v, false, bf.v,
                                                       (short)0, acc[nt],
                                                       false, false);
    }
  }

  const int mbase = tM * 16 + half_sel * 8;
#pragma unroll
  for (int nt = 0; nt < 4; ++nt) {
    int n_glob = tN * 64 + nt * 16 + l15;
    float bn = bias ? bias[n_glob] : 0.0f;
#pragma unroll
    for (int v = 0; v < 8; ++v) {
      int m = mbase + v;
      if (m < R) {
        float val = acc[nt][v] + bn;
        if (relu) val = fmaxf(val, 0.0f);
        if (C32) C32[(size_t)m * ldc + n_glob] = val;
        if (C16) C16[(size_t)m * ldc + n_glob] = (h16)val;
      }
    }
  }
}

// ---------------------------------------------------------------------------
// Max over K grouped rows: in f16 [B*S*K, ldc] -> out f32 [B,S,C].
// ---------------------------------------------------------------------------
__global__ void k_maxpool(const h16* __restrict__ in, int ldc, int S, int K,
                          int C, int Btot, float* __restrict__ out) {
  int t = blockIdx.x * blockDim.x + threadIdx.x;
  int total = Btot * S * C;
  if (t >= total) return;
  int c = t % C;
  int s = (t / C) % S;
  int b = t / (C * S);
  const h16* base = in + ((size_t)(b * S + s) * K) * ldc + c;
  float m = -3e38f;
  for (int k = 0; k < K; ++k) {
    float v = (float)base[(size_t)k * ldc];
    m = fmaxf(m, v);
  }
  out[t] = m;
}

// ---------------------------------------------------------------------------
// sa3 (group_all) feature build: [l2_xyz(3) | l2_points(256) | pad->288]
// ---------------------------------------------------------------------------
__global__ void k_feat_sa3(const float* __restrict__ l2xyz,
                           const float* __restrict__ l2pts,
                           h16* __restrict__ out, int ldk, int total) {
  int r = blockIdx.x * blockDim.x + threadIdx.x;
  if (r >= total) return;
  h16* o = out + (size_t)r * ldk;
  const float* xr = l2xyz + (size_t)r * 3;
  o[0] = (h16)xr[0]; o[1] = (h16)xr[1]; o[2] = (h16)xr[2];
  const float* pr = l2pts + (size_t)r * 256;
  for (int c = 0; c < 256; ++c) o[3 + c] = (h16)pr[c];
  for (int c = 259; c < ldk; ++c) o[c] = (h16)0.0f;
}

// ---------------------------------------------------------------------------
// up3 feature build: [l2_points(256) | broadcast l3_points(1024)]
// ---------------------------------------------------------------------------
__global__ void k_feat_up3(const float* __restrict__ l2pts,
                           const float* __restrict__ l3pts,
                           h16* __restrict__ out, int total) {
  int r = blockIdx.x * blockDim.x + threadIdx.x;
  if (r >= total) return;
  int b = r >> 7;  // 128 rows per batch
  h16* o = out + (size_t)r * 1280;
  const float* pr = l2pts + (size_t)r * 256;
  for (int c = 0; c < 256; ++c) o[c] = (h16)pr[c];
  const float* g = l3pts + (size_t)b * 1024;
  for (int c = 0; c < 1024; ++c) o[256 + c] = (h16)g[c];
}

// ---------------------------------------------------------------------------
// Attention core: one block per (b, head, query row). L=128, dk=dv=64, C=256.
// ---------------------------------------------------------------------------
__global__ void k_attn(const h16* __restrict__ q, const h16* __restrict__ k,
                       const h16* __restrict__ v, h16* __restrict__ o) {
  const int L = 128;
  int qi = blockIdx.x % L;
  int h = (blockIdx.x / L) % 4;
  int b = blockIdx.x / (L * 4);
  int tid = threadIdx.x;  // 0..127

  __shared__ float sc[128];
  __shared__ float red[128];

  const h16* qrow = q + ((size_t)(b * L + qi) * 256) + h * 64;
  const h16* krow = k + ((size_t)(b * L + tid) * 256) + h * 64;
  float s = 0.0f;
  for (int d = 0; d < 64; ++d) s += (float)qrow[d] * (float)krow[d];
  s *= 0.125f;  // 1/sqrt(64)

  red[tid] = s;
  __syncthreads();
  for (int st = 64; st > 0; st >>= 1) {
    if (tid < st) red[tid] = fmaxf(red[tid], red[tid + st]);
    __syncthreads();
  }
  float mx = red[0];
  __syncthreads();
  float e = expf(s - mx);
  red[tid] = e;
  __syncthreads();
  for (int st = 64; st > 0; st >>= 1) {
    if (tid < st) red[tid] += red[tid + st];
    __syncthreads();
  }
  float denom = red[0];
  __syncthreads();
  sc[tid] = e / denom;
  __syncthreads();

  if (tid < 64) {
    float accv = 0.0f;
    const h16* vbase = v + ((size_t)(b * L) * 256) + h * 64 + tid;
    for (int j = 0; j < L; ++j) accv += sc[j] * (float)vbase[(size_t)j * 256];
    o[((size_t)(b * L + qi) * 256) + h * 64 + tid] = (h16)accv;
  }
}

// ---------------------------------------------------------------------------
// Residual add + LayerNorm (no affine), C=256. Emits f32 + f16.
// ---------------------------------------------------------------------------
__global__ void k_addln(const float* __restrict__ a, const float* __restrict__ bb,
                        float* __restrict__ o32, h16* __restrict__ o16) {
  int row = blockIdx.x;
  int t = threadIdx.x;  // 256 threads
  __shared__ float red[256];
  size_t i = (size_t)row * 256 + t;
  float vv = a[i] + bb[i];
  red[t] = vv;
  __syncthreads();
  for (int s = 128; s > 0; s >>= 1) {
    if (t < s) red[t] += red[t + s];
    __syncthreads();
  }
  float mean = red[0] * (1.0f / 256.0f);
  __syncthreads();
  float d = vv - mean;
  red[t] = d * d;
  __syncthreads();
  for (int s = 128; s > 0; s >>= 1) {
    if (t < s) red[t] += red[t + s];
    __syncthreads();
  }
  float var = red[0] * (1.0f / 256.0f);
  float y = d * rsqrtf(var + 1e-5f);
  o32[i] = y;
  o16[i] = (h16)y;
}

// ---------------------------------------------------------------------------
// 3-NN with inverse-distance weights.
// ---------------------------------------------------------------------------
__global__ void k_knn3(const float* __restrict__ xyz1, const float* __restrict__ xyz2,
                       int n1, int n2, int* __restrict__ i3, float* __restrict__ w3,
                       int total) {
  int t = blockIdx.x * blockDim.x + threadIdx.x;
  if (t >= total) return;
  int b = t / n1;
  const float* p = xyz1 + (size_t)t * 3;
  float px = p[0], py = p[1], pz = p[2];
  const float* qq = xyz2 + (size_t)b * n2 * 3;
  float d0 = 3e38f, d1 = 3e38f, d2 = 3e38f;
  int j0 = 0, j1 = 0, j2 = 0;
  for (int j = 0; j < n2; ++j) {
    float dx = qq[j * 3 + 0] - px, dy = qq[j * 3 + 1] - py, dz = qq[j * 3 + 2] - pz;
    float d = dx * dx + dy * dy + dz * dz;
    if (d < d0)      { d2 = d1; j2 = j1; d1 = d0; j1 = j0; d0 = d; j0 = j; }
    else if (d < d1) { d2 = d1; j2 = j1; d1 = d;  j1 = j; }
    else if (d < d2) { d2 = d;  j2 = j; }
  }
  float w0 = 1.0f / (d0 + 1e-8f), w1 = 1.0f / (d1 + 1e-8f), w2 = 1.0f / (d2 + 1e-8f);
  float ws = w0 + w1 + w2;
  w0 /= ws; w1 /= ws; w2 /= ws;
  i3[t * 3 + 0] = j0; i3[t * 3 + 1] = j1; i3[t * 3 + 2] = j2;
  w3[t * 3 + 0] = w0; w3[t * 3 + 1] = w1; w3[t * 3 + 2] = w2;
}

// ---------------------------------------------------------------------------
// up2 feature build: [l1_points(128) | interp(l2pB, 256)]; ldk = 384.
// ---------------------------------------------------------------------------
__global__ void k_feat_up2(const float* __restrict__ l1pts,
                           const float* __restrict__ l2p,
                           const int* __restrict__ i3, const float* __restrict__ w3,
                           h16* __restrict__ out, int total) {
  int r = blockIdx.x * blockDim.x + threadIdx.x;
  if (r >= total) return;
  int b = r / 358;
  h16* o = out + (size_t)r * 384;
  const float* f1 = l1pts + (size_t)r * 128;
  for (int c = 0; c < 128; ++c) o[c] = (h16)f1[c];
  int j0 = i3[r * 3], j1 = i3[r * 3 + 1], j2 = i3[r * 3 + 2];
  float w0 = w3[r * 3], w1 = w3[r * 3 + 1], w2 = w3[r * 3 + 2];
  const float* base = l2p + (size_t)b * 128 * 256;
  for (int c = 0; c < 256; ++c) {
    float s = w0 * base[(size_t)j0 * 256 + c] + w1 * base[(size_t)j1 * 256 + c] +
              w2 * base[(size_t)j2 * 256 + c];
    o[128 + c] = (h16)s;
  }
}

// ---------------------------------------------------------------------------
// up1 feature build: [xyz(3) | xyz(3) | interp(l1pB,128) | pad->160].
// ---------------------------------------------------------------------------
__global__ void k_feat_up1(const float* __restrict__ x,
                           const float* __restrict__ l1p,
                           const int* __restrict__ i3, const float* __restrict__ w3,
                           h16* __restrict__ out, int total) {
  int r = blockIdx.x * blockDim.x + threadIdx.x;
  if (r >= total) return;
  int b = r >> 14;  // 16384 points per batch
  h16* o = out + (size_t)r * 160;
  const float* xp = x + (size_t)r * 3;
  o[0] = (h16)xp[0]; o[1] = (h16)xp[1]; o[2] = (h16)xp[2];
  o[3] = (h16)xp[0]; o[4] = (h16)xp[1]; o[5] = (h16)xp[2];
  int j0 = i3[r * 3], j1 = i3[r * 3 + 1], j2 = i3[r * 3 + 2];
  float w0 = w3[r * 3], w1 = w3[r * 3 + 1], w2 = w3[r * 3 + 2];
  const float* base = l1p + (size_t)b * 358 * 128;
  for (int c = 0; c < 128; ++c) {
    float s = w0 * base[(size_t)j0 * 128 + c] + w1 * base[(size_t)j1 * 128 + c] +
              w2 * base[(size_t)j2 * 128 + c];
    o[6 + c] = (h16)s;
  }
  for (int c = 134; c < 160; ++c) o[c] = (h16)0.0f;
}

// ===========================================================================
extern "C" void kernel_launch(void* const* d_in, const int* in_sizes, int n_in,
                              void* d_out, int out_size, void* d_ws, size_t ws_size,
                              hipStream_t stream) {
  (void)in_sizes; (void)n_in; (void)out_size; (void)ws_size;

  // --- inputs (jax tree-flatten order: dict keys sorted, lists in order) ---
  const float* X     = (const float*)d_in[0];
  const float* aw1   = (const float*)d_in[1];   // att.w1  [1024,256]
  const float* aw2   = (const float*)d_in[2];   // att.w2  [256,1024]
  const float* awk   = (const float*)d_in[3];   // att.wk  [256,256]
  const float* awo   = (const float*)d_in[4];   // att.wo
  const float* awq   = (const float*)d_in[5];   // att.wq
  const float* awv   = (const float*)d_in[6];   // att.wv
  const float* ds1b[3] = {(const float*)d_in[7],  (const float*)d_in[8],  (const float*)d_in[9]};
  const float* ds1w[3] = {(const float*)d_in[10], (const float*)d_in[11], (const float*)d_in[12]};
  const float* ds2b[3] = {(const float*)d_in[13], (const float*)d_in[14], (const float*)d_in[15]};
  const float* ds2w[3] = {(const float*)d_in[16], (const float*)d_in[17], (const float*)d_in[18]};
  const float* ds3b[3] = {(const float*)d_in[19], (const float*)d_in[20], (const float*)d_in[21]};
  const float* ds3w[3] = {(const float*)d_in[22], (const float*)d_in[23], (const float*)d_in[24]};
  const float* up1b[3] = {(const float*)d_in[25], (const float*)d_in[26], (const float*)d_in[27]};
  const float* up1w[3] = {(const float*)d_in[28], (const float*)d_in[29], (const float*)d_in[30]};
  const float* up2b[2] = {(const float*)d_in[31], (const float*)d_in[32]};
  const float* up2w[2] = {(const float*)d_in[33], (const float*)d_in[34]};
  const float* up3b[2] = {(const float*)d_in[35], (const float*)d_in[36]};
  const float* up3w[2] = {(const float*)d_in[37], (const float*)d_in[38]};

  const int Bc = 8, Nc = 16384;
  const int S1 = 358, K1 = 64, R1 = Bc * S1 * K1;   // 183296
  const int S2 = 128, K2 = 64, R2 = Bc * S2 * K2;   // 65536
  const int Rm = Bc * 128;                          // 1024
  const int Ru2 = Bc * S1;                          // 2864
  const int Ru1 = Bc * Nc;                          // 131072

  // --- bump allocator over d_ws ---
  char* ws = (char*)d_ws;
  size_t off = 0;
  auto alloc = [&](size_t bytes) -> void* {
    void* p = ws + off;
    off = (off + bytes + 255) & ~(size_t)255;
    return p;
  };
  auto allocH = [&](size_t n) -> h16*  { return (h16*)alloc(n * sizeof(h16)); };
  auto allocF = [&](size_t n) -> float*{ return (float*)alloc(n * sizeof(float)); };
  auto allocI = [&](size_t n) -> int*  { return (int*)alloc(n * sizeof(int)); };

  auto wpad = [&](const float* w, int Nr, int K, int Kp) -> h16* {
    h16* o = allocH((size_t)Nr * Kp);
    int tot = Nr * Kp;
    k_wpad<<<(tot + 255) / 256, 256, 0, stream>>>(w, o, Nr, K, Kp);
    return o;
  };

  // N always a multiple of 64 here -> 16x64 tiles (grid.y = N/64).
  auto gemm = [&](const h16* A, int Kp, const h16* Wp, const float* bias, int Nout,
                  float* C32, h16* C16, int R, int relu) {
    dim3 g((unsigned)((R + 15) / 16), (unsigned)(Nout / 64));
    switch (Kp) {
      case 32:   k_gemm<32><<<g, 32, 0, stream>>>(A, Wp, bias, C32, C16, Nout, R, relu); break;
      case 64:   k_gemm<64><<<g, 32, 0, stream>>>(A, Wp, bias, C32, C16, Nout, R, relu); break;
      case 128:  k_gemm<128><<<g, 32, 0, stream>>>(A, Wp, bias, C32, C16, Nout, R, relu); break;
      case 160:  k_gemm<160><<<g, 32, 0, stream>>>(A, Wp, bias, C32, C16, Nout, R, relu); break;
      case 256:  k_gemm<256><<<g, 32, 0, stream>>>(A, Wp, bias, C32, C16, Nout, R, relu); break;
      case 288:  k_gemm<288><<<g, 32, 0, stream>>>(A, Wp, bias, C32, C16, Nout, R, relu); break;
      case 384:  k_gemm<384><<<g, 32, 0, stream>>>(A, Wp, bias, C32, C16, Nout, R, relu); break;
      case 512:  k_gemm<512><<<g, 32, 0, stream>>>(A, Wp, bias, C32, C16, Nout, R, relu); break;
      case 1024: k_gemm<1024><<<g, 32, 0, stream>>>(A, Wp, bias, C32, C16, Nout, R, relu); break;
      case 1280: k_gemm<1280><<<g, 32, 0, stream>>>(A, Wp, bias, C32, C16, Nout, R, relu); break;
      default: break;
    }
  };

  // --- f16 weight conversions (K padded to multiple of 32; N already %64) ---
  h16* w_ds1[3] = {wpad(ds1w[0], 64, 6, 32),    wpad(ds1w[1], 64, 64, 64),   wpad(ds1w[2], 128, 64, 64)};
  h16* w_ds2[3] = {wpad(ds2w[0], 128, 131, 160), wpad(ds2w[1], 128, 128, 128), wpad(ds2w[2], 256, 128, 128)};
  h16* w_ds3[3] = {wpad(ds3w[0], 256, 259, 288), wpad(ds3w[1], 512, 256, 256), wpad(ds3w[2], 1024, 512, 512)};
  h16* w_up3[2] = {wpad(up3w[0], 256, 1280, 1280), wpad(up3w[1], 256, 256, 256)};
  h16* w_q  = wpad(awq, 256, 256, 256);
  h16* w_k  = wpad(awk, 256, 256, 256);
  h16* w_v  = wpad(awv, 256, 256, 256);
  h16* w_o  = wpad(awo, 256, 256, 256);
  h16* w_f1 = wpad(aw1, 1024, 256, 256);
  h16* w_f2 = wpad(aw2, 256, 1024, 1024);
  h16* w_up2[2] = {wpad(up2w[0], 256, 384, 384), wpad(up2w[1], 128, 256, 256)};
  h16* w_up1[3] = {wpad(up1w[0], 512, 134, 160), wpad(up1w[1], 256, 512, 512), wpad(up1w[2], 256, 256, 256)};

  // --- stage buffers ---
  float* l1_xyz = allocF((size_t)Bc * S1 * 3);
  float* l2_xyz = allocF((size_t)Bc * S2 * 3);
  int*   gidx1  = allocI((size_t)R1);
  int*   gidx2  = allocI((size_t)R2);
  float* l1_pts = allocF((size_t)Bc * S1 * 128);
  float* l2_pts = allocF((size_t)Bc * S2 * 256);
  float* l3_pts = allocF((size_t)Bc * 1024);

  h16* feat1 = allocH((size_t)R1 * 32);
  h16* h1a   = allocH((size_t)R1 * 64);
  h16* h1b   = allocH((size_t)R1 * 64);
  h16* h1c   = allocH((size_t)R1 * 128);

  h16* feat2 = allocH((size_t)R2 * 160);
  h16* h2a   = allocH((size_t)R2 * 128);
  h16* h2b   = allocH((size_t)R2 * 128);
  h16* h2c   = allocH((size_t)R2 * 256);

  h16* feat3 = allocH((size_t)Rm * 288);
  h16* h3a   = allocH((size_t)Rm * 256);
  h16* h3b   = allocH((size_t)Rm * 512);
  h16* h3c   = allocH((size_t)Rm * 1024);

  h16*   featU3 = allocH((size_t)Rm * 1280);
  h16*   hU3    = allocH((size_t)Rm * 256);
  float* l2pA32 = allocF((size_t)Rm * 256);
  h16*   l2pA16 = allocH((size_t)Rm * 256);

  h16*   q16    = allocH((size_t)Rm * 256);
  h16*   k16    = allocH((size_t)Rm * 256);
  h16*   v16    = allocH((size_t)Rm * 256);
  h16*   attn16 = allocH((size_t)Rm * 256);
  float* opre32 = allocF((size_t)Rm * 256);
  float* o32    = allocF((size_t)Rm * 256);
  h16*   o16    = allocH((size_t)Rm * 256);
  h16*   hff    = allocH((size_t)Rm * 1024);
  float* ff232  = allocF((size_t)Rm * 256);
  float* l2pB32 = allocF((size_t)Rm * 256);
  h16*   l2pB16 = allocH((size_t)Rm * 256);

  int*   i3u2 = allocI((size_t)Ru2 * 3);
  float* w3u2 = allocF((size_t)Ru2 * 3);
  h16*   featU2 = allocH((size_t)Ru2 * 384);
  h16*   hU2    = allocH((size_t)Ru2 * 256);
  float* l1pB32 = allocF((size_t)Ru2 * 128);

  int*   i3u1 = allocI((size_t)Ru1 * 3);
  float* w3u1 = allocF((size_t)Ru1 * 3);
  h16*   featU1 = allocH((size_t)Ru1 * 160);
  h16*   hU1a   = allocH((size_t)Ru1 * 512);
  h16*   hU1b   = allocH((size_t)Ru1 * 256);

  // ======================= SA1 =======================
  k_fps<16><<<Bc, 1024, 0, stream>>>(X, Nc, S1, l1_xyz);
  k_ballquery<<<Bc * S1, 32, 0, stream>>>(X, l1_xyz, gidx1, Nc, S1, K1, 0.04f);
  k_group_feat<<<(R1 + 255) / 256, 256, 0, stream>>>(X, Nc, X, Nc, 3, l1_xyz, gidx1,
                                                     S1, K1, feat1, 32, R1);
  gemm(feat1, 32, w_ds1[0], ds1b[0], 64, nullptr, h1a, R1, 1);
  gemm(h1a, 64, w_ds1[1], ds1b[1], 64, nullptr, h1b, R1, 1);
  gemm(h1b, 64, w_ds1[2], ds1b[2], 128, nullptr, h1c, R1, 1);
  {
    int tot = Bc * S1 * 128;
    k_maxpool<<<(tot + 255) / 256, 256, 0, stream>>>(h1c, 128, S1, K1, 128, Bc, l1_pts);
  }

  // ======================= SA2 =======================
  k_fps<1><<<Bc, 512, 0, stream>>>(l1_xyz, S1, S2, l2_xyz);
  k_ballquery<<<Bc * S2, 32, 0, stream>>>(l1_xyz, l2_xyz, gidx2, S1, S2, K2, 0.16f);
  k_group_feat<<<(R2 + 255) / 256, 256, 0, stream>>>(l1_xyz, S1, l1_pts, S1, 128,
                                                     l2_xyz, gidx2, S2, K2, feat2, 160, R2);
  gemm(feat2, 160, w_ds2[0], ds2b[0], 128, nullptr, h2a, R2, 1);
  gemm(h2a, 128, w_ds2[1], ds2b[1], 128, nullptr, h2b, R2, 1);
  gemm(h2b, 128, w_ds2[2], ds2b[2], 256, nullptr, h2c, R2, 1);
  {
    int tot = Bc * S2 * 256;
    k_maxpool<<<(tot + 255) / 256, 256, 0, stream>>>(h2c, 256, S2, K2, 256, Bc, l2_pts);
  }

  // ======================= SA3 (group_all) =======================
  k_feat_sa3<<<(Rm + 255) / 256, 256, 0, stream>>>(l2_xyz, l2_pts, feat3, 288, Rm);
  gemm(feat3, 288, w_ds3[0], ds3b[0], 256, nullptr, h3a, Rm, 1);
  gemm(h3a, 256, w_ds3[1], ds3b[1], 512, nullptr, h3b, Rm, 1);
  gemm(h3b, 512, w_ds3[2], ds3b[2], 1024, nullptr, h3c, Rm, 1);
  {
    int tot = Bc * 1024;
    k_maxpool<<<(tot + 255) / 256, 256, 0, stream>>>(h3c, 1024, 1, 128, 1024, Bc, l3_pts);
  }

  // ======================= FP up3 =======================
  k_feat_up3<<<(Rm + 255) / 256, 256, 0, stream>>>(l2_pts, l3_pts, featU3, Rm);
  gemm(featU3, 1280, w_up3[0], up3b[0], 256, nullptr, hU3, Rm, 1);
  gemm(hU3, 256, w_up3[1], up3b[1], 256, l2pA32, l2pA16, Rm, 1);

  // ======================= MHA =======================
  gemm(l2pA16, 256, w_q, nullptr, 256, nullptr, q16, Rm, 0);
  gemm(l2pA16, 256, w_k, nullptr, 256, nullptr, k16, Rm, 0);
  gemm(l2pA16, 256, w_v, nullptr, 256, nullptr, v16, Rm, 0);
  k_attn<<<Bc * 4 * 128, 128, 0, stream>>>(q16, k16, v16, attn16);
  gemm(attn16, 256, w_o, nullptr, 256, opre32, nullptr, Rm, 0);
  k_addln<<<Rm, 256, 0, stream>>>(l2pA32, opre32, o32, o16);
  gemm(o16, 256, w_f1, nullptr, 1024, nullptr, hff, Rm, 1);
  gemm(hff, 1024, w_f2, nullptr, 256, ff232, nullptr, Rm, 0);
  k_addln<<<Rm, 256, 0, stream>>>(o32, ff232, l2pB32, l2pB16);

  // ======================= FP up2 =======================
  k_knn3<<<(Ru2 + 127) / 128, 128, 0, stream>>>(l1_xyz, l2_xyz, S1, S2, i3u2, w3u2, Ru2);
  k_feat_up2<<<(Ru2 + 127) / 128, 128, 0, stream>>>(l1_pts, l2pB32, i3u2, w3u2, featU2, Ru2);
  gemm(featU2, 384, w_up2[0], up2b[0], 256, nullptr, hU2, Ru2, 1);
  gemm(hU2, 256, w_up2[1], up2b[1], 128, l1pB32, nullptr, Ru2, 1);

  // ======================= FP up1 =======================
  k_knn3<<<(Ru1 + 127) / 128, 128, 0, stream>>>(X, l1_xyz, Nc, S1, i3u1, w3u1, Ru1);
  k_feat_up1<<<(Ru1 + 127) / 128, 128, 0, stream>>>(X, l1pB32, i3u1, w3u1, featU1, Ru1);
  gemm(featU1, 160, w_up1[0], up1b[0], 512, nullptr, hU1a, Ru1, 1);
  gemm(hU1a, 512, w_up1[1], up1b[1], 256, nullptr, hU1b, Ru1, 1);
  gemm(hU1b, 256, w_up1[2], up1b[2], 256, (float*)d_out, nullptr, Ru1, 1);
}
